// MultiHeadAttention_74165495267625
// MI455X (gfx1250) — matrix operand order repaired and checked by reading, not compiled, not run on hardware
//
#include <hip/hip_runtime.h>

typedef __attribute__((ext_vector_type(16))) _Float16 v16h;
typedef __attribute__((ext_vector_type(8)))  _Float16 v8h;
typedef __attribute__((ext_vector_type(8)))  float    v8f;
typedef __attribute__((ext_vector_type(4)))  unsigned int v4u;
typedef __attribute__((ext_vector_type(8)))  int v8i;
typedef __attribute__((ext_vector_type(4)))  int v4i;

#ifndef __has_builtin
#define __has_builtin(x) 0
#endif
#if __has_builtin(__builtin_amdgcn_tensor_load_to_lds)
#define HAVE_TDM 1
#else
#define HAVE_TDM 0
#endif

// ---------------------------------------------------------------------------
// WMMA helper: D = A(16x32 f16) x B(32x16 f16) + C(16x16 f32)
// ---------------------------------------------------------------------------
__device__ __forceinline__ v8f wmma32(v16h a, v16h b, v8f c) {
  return __builtin_amdgcn_wmma_f32_16x16x32_f16(
      /*neg_a=*/false, a, /*neg_b=*/false, b,
      /*c_mod=*/(short)0, c, /*reuse_a=*/false, /*reuse_b=*/false);
}

// A fragment (16x32, MxK) from a row-major [*, ld] f16 source.
// lane = 16*h + r holds row m0+r, K = {k0+8h..+7} U {k0+16+8h..+7}
__device__ __forceinline__ v16h load_frag_a(const _Float16* base, int ld, int m0, int k0) {
  const int lane = threadIdx.x & 31;
  const int r = lane & 15, h = lane >> 4;
  const _Float16* p = base + (size_t)(m0 + r) * ld + k0 + 8 * h;
  v8h lo = *(const v8h*)p;
  v8h hi = *(const v8h*)(p + 16);
  v16h out;
#pragma unroll
  for (int i = 0; i < 8; ++i) { out[i] = lo[i]; out[8 + i] = hi[i]; }
  return out;
}

// B fragment (32x16, KxN): column n of B = row (n0+n) of a row-major source.
// lane = 16*h + n holds column n0+n, K = {k0+8h..+7} U {k0+16+8h..+7}
__device__ __forceinline__ v16h load_frag_b(const _Float16* base, int ld, int n0, int k0) {
  const int lane = threadIdx.x & 31;
  const int r = lane & 15, h = lane >> 4;
  const _Float16* p = base + (size_t)(n0 + r) * ld + k0 + 8 * h;
  v8h lo = *(const v8h*)p;
  v8h hi = *(const v8h*)(p + 16);
  v16h out;
#pragma unroll
  for (int i = 0; i < 8; ++i) { out[i] = lo[i]; out[8 + i] = hi[i]; }
  return out;
}

// ---------------------------------------------------------------------------
// TDM: 2D tensor-tile load (global -> LDS).  Dims/strides in elements
// (data_size = 2 bytes), global_addr in bytes, lds_addr = LDS byte offset.
// ---------------------------------------------------------------------------
#if HAVE_TDM
__device__ __forceinline__ void tdm_load_2d(unsigned lds_addr, const void* gptr,
                                            unsigned td0, unsigned td1,
                                            unsigned tile0, unsigned tile1,
                                            unsigned long long stride0) {
  const unsigned long long ga = (unsigned long long)(uintptr_t)gptr;
  v4u g0;
  g0[0] = 1u;                                    // count=1, user descriptor
  g0[1] = lds_addr;                              // lds_addr [63:32]
  g0[2] = (unsigned)(ga & 0xFFFFFFFFu);          // global_addr [95:64]
  g0[3] = (unsigned)((ga >> 32) & 0x01FFFFFFu)   // global_addr [120:96]
        | (2u << 30);                            // type = 2 ("image")
  v8i g1;
  g1[0] = (int)(1u << 16);                                            // data_size=2B
  g1[1] = (int)((td0 & 0xFFFFu) << 16);                               // tensor_dim0 lo
  g1[2] = (int)(((td0 >> 16) & 0xFFFFu) | ((td1 & 0xFFFFu) << 16));   // td0 hi | td1 lo
  g1[3] = (int)(((td1 >> 16) & 0xFFFFu) | ((tile0 & 0xFFFFu) << 16)); // td1 hi | tile0
  g1[4] = (int)(tile1 & 0xFFFFu);                                     // tile1 | tile2=0
  g1[5] = (int)(unsigned)(stride0 & 0xFFFFFFFFu);                     // dim0 stride lo
  g1[6] = (int)((stride0 >> 32) & 0xFFFFu);                           // stride hi | s1=0
  g1[7] = 0;
  v4i z4 = {0, 0, 0, 0};
  v8i z8 = {0, 0, 0, 0, 0, 0, 0, 0};
  __builtin_amdgcn_tensor_load_to_lds(g0, g1, z4, z4, z8, 0);
}
#endif

__device__ __forceinline__ void tdm_wait() {
#if __has_builtin(__builtin_amdgcn_s_wait_tensorcnt)
  __builtin_amdgcn_s_wait_tensorcnt(0);
#else
  asm volatile("s_wait_tensorcnt 0x0" ::: "memory");
#endif
}

// ---------------------------------------------------------------------------
// fp32 -> fp16 conversion
// ---------------------------------------------------------------------------
__global__ void cvt_f32_f16(const float* __restrict__ x, _Float16* __restrict__ y, int n) {
  for (int i = blockIdx.x * blockDim.x + threadIdx.x; i < n; i += gridDim.x * blockDim.x)
    y[i] = (_Float16)x[i];
}

// ---------------------------------------------------------------------------
// GEMM: C[M,N] = A[M,K] (f16 row-major) x W[N,K]^T (f16 row-major).
// Block tile 64(M) x 128(N), k-step 32; 128 threads = 4 waves (2x2), 32x64 each.
// A/W tiles staged in LDS by the TDM, double-buffered: DMA of chunk k+1
// overlaps the 8 WMMAs of chunk k; wave 0 waits TENSORcnt, then barrier.
// MODE 0: f16 out; MODE 1: f16 transposed out (Ch[n*M+m], packed 16B stores);
// MODE 2: f32 out.
// ---------------------------------------------------------------------------
template <int MODE>
__global__ void __launch_bounds__(128)
gemm_nt(const _Float16* __restrict__ A, const _Float16* __restrict__ W,
        _Float16* __restrict__ Ch, float* __restrict__ Cf, int M, int N, int K) {
  const int wave = threadIdx.x >> 5;
  const int wm = wave >> 1, wn = wave & 1;
  const int mblk = blockIdx.y * 64;
  const int nblk = blockIdx.x * 128;
  const int m0 = wm * 32;   // wave-local tile origin
  const int n0 = wn * 64;

  __shared__ __align__(32) _Float16 As[2][64][32];    //  8 KB
  __shared__ __align__(32) _Float16 Ws[2][128][32];   // 16 KB

  auto stage = [&](int buf, int k0) {
#if HAVE_TDM
    if (wave == 0) {
      // A tile: 64 rows x 32 k  |  W tile: 128 rows x 32 k (k contiguous)
      tdm_load_2d((unsigned)(uintptr_t)&As[buf][0][0], A + (size_t)mblk * K + k0,
                  /*td0=*/(unsigned)K, /*td1=*/(unsigned)M, 32, 64,
                  (unsigned long long)K);
      tdm_load_2d((unsigned)(uintptr_t)&Ws[buf][0][0], W + (size_t)nblk * K + k0,
                  /*td0=*/(unsigned)K, /*td1=*/(unsigned)N, 32, 128,
                  (unsigned long long)K);
    }
#else
    {
      const int ra = threadIdx.x >> 1;            // 0..63
      const int ca = (threadIdx.x & 1) * 16;
      *(v16h*)&As[buf][ra][ca] = *(const v16h*)(A + (size_t)(mblk + ra) * K + k0 + ca);
      const int rw = threadIdx.x;                 // 0..127
      const _Float16* wsrc = W + (size_t)(nblk + rw) * K + k0;
      *(v16h*)&Ws[buf][rw][0]  = *(const v16h*)wsrc;
      *(v16h*)&Ws[buf][rw][16] = *(const v16h*)(wsrc + 16);
    }
#endif
  };

  v8f acc[2][4] = {};

  stage(0, 0);
#if HAVE_TDM
  if (wave == 0) tdm_wait();
#endif
  __syncthreads();

  int buf = 0;
  for (int k0 = 0; k0 < K; k0 += 32) {
    if (k0 + 32 < K) stage(buf ^ 1, k0 + 32);   // DMA next chunk during compute

    v16h a0 = load_frag_a(&As[buf][0][0], 32, m0, 0);
    v16h a1 = load_frag_a(&As[buf][0][0], 32, m0 + 16, 0);
#pragma unroll
    for (int j = 0; j < 4; ++j) {
      v16h b = load_frag_b(&Ws[buf][0][0], 32, n0 + j * 16, 0);
      acc[0][j] = wmma32(a0, b, acc[0][j]);
      acc[1][j] = wmma32(a1, b, acc[1][j]);
    }

#if HAVE_TDM
    if (wave == 0) tdm_wait();                  // next tile landed
#endif
    __syncthreads();
    buf ^= 1;
  }

  const int lane = threadIdx.x & 31;
  const int col = lane & 15, rs = lane >> 4;
#pragma unroll
  for (int i = 0; i < 2; ++i)
#pragma unroll
    for (int j = 0; j < 4; ++j) {
      if (MODE == 1) {  // transposed: per-lane m-run is contiguous -> 16B store
        v8h pk;
#pragma unroll
        for (int v = 0; v < 8; ++v) pk[v] = (_Float16)acc[i][j][v];
        const int n = nblk + n0 + j * 16 + col;
        *(v8h*)&Ch[(size_t)n * M + mblk + m0 + i * 16 + 8 * rs] = pk;
      } else {
#pragma unroll
        for (int v = 0; v < 8; ++v) {
          const int m = mblk + m0 + i * 16 + v + 8 * rs;
          const int n = nblk + n0 + j * 16 + col;
          if (MODE == 2) Cf[(size_t)m * N + n] = acc[i][j][v];
          else           Ch[(size_t)m * N + n] = (_Float16)acc[i][j][v];
        }
      }
    }
}

// ---------------------------------------------------------------------------
// Causal GQA flash attention.
//   Qp [2048, 32*128] f16, Kp [2048, 8*128] f16, VpT [8*128, 2048] f16 (V^T!)
//   O  [2048, 32*128] f16
// grid = (qblocks=32, heads=32); 128 threads = 4 waves; wave owns 16 q rows.
// K / V^T tiles double-buffered in LDS via TDM; S^T = K @ Q^T via WMMA,
// online softmax (one shfl_xor(16)), P via per-wave LDS (b128 stores),
// O^T += V^T @ P^T via WMMA.
// ---------------------------------------------------------------------------
__global__ void __launch_bounds__(128)
attn_fwd(const _Float16* __restrict__ Qp, const _Float16* __restrict__ Kp,
         const _Float16* __restrict__ VpT, _Float16* __restrict__ O) {
  const int head = blockIdx.y;
  const int kvh  = head >> 2;           // GQA: 4 query heads per KV head
  const int qblk = blockIdx.x;
  const int wave = threadIdx.x >> 5;
  const int lane = threadIdx.x & 31;
  const int q0   = qblk * 64 + wave * 16;
  const int ldq = 4096, ldkv = 1024, ldvt = 2048;

  const _Float16* Qh  = Qp  + (size_t)head * 128;
  const _Float16* Kh  = Kp  + (size_t)kvh * 128;
  const _Float16* VTh = VpT + (size_t)kvh * 128 * ldvt;   // this head's 128 dv rows

  __shared__ __align__(32) _Float16 Ks[2][32][128];   // K tiles   [k][d]  16 KB
  __shared__ __align__(32) _Float16 Vts[2][128][32];  // V^T tiles [dv][k] 16 KB
  __shared__ __align__(32) _Float16 Ps[4][16][32];    // per-wave P [q][k]  4 KB

  auto stage = [&](int buf, int kc) {
#if HAVE_TDM
    if (wave == 0) {
      // K tile: tensor [2048 x 1024], tile 32 rows x 128 cols
      tdm_load_2d((unsigned)(uintptr_t)&Ks[buf][0][0], Kh + (size_t)kc * ldkv,
                  /*td0=*/ldkv, /*td1=*/2048, 128, 32, (unsigned long long)ldkv);
      // V^T tile: tensor [1024 x 2048], tile 128 rows x 32 cols
      tdm_load_2d((unsigned)(uintptr_t)&Vts[buf][0][0], VTh + kc,
                  /*td0=*/ldvt, /*td1=*/1024, 32, 128, (unsigned long long)ldvt);
    }
#else
    {
      const int r = threadIdx.x >> 2;
      const int c = (threadIdx.x & 3) * 32;
      const _Float16* ksrc = Kh + (size_t)(kc + r) * ldkv + c;
      *(v16h*)&Ks[buf][r][c]      = *(const v16h*)ksrc;
      *(v16h*)&Ks[buf][r][c + 16] = *(const v16h*)(ksrc + 16);
      const int rv = threadIdx.x;  // 0..127 -> one dv row each
      const _Float16* vsrc = VTh + (size_t)rv * ldvt + kc;
      *(v16h*)&Vts[buf][rv][0]  = *(const v16h*)vsrc;
      *(v16h*)&Vts[buf][rv][16] = *(const v16h*)(vsrc + 16);
    }
#endif
  };

  // Q fragments (B operand of S^T): constant across the kv loop
  v16h qb[4];
#pragma unroll
  for (int c = 0; c < 4; ++c) qb[c] = load_frag_b(Qh, ldq, q0, c * 32);

  v8f ot[8] = {};                       // O^T accum: tile i = dv rows [16i,16i+16)
  float m_run = -1e30f, l_run = 0.0f;
  const float scale = 0.08838834764831845f;  // 1/sqrt(128)

  const int kend = qblk * 64 + 64;

  stage(0, 0);
#if HAVE_TDM
  if (wave == 0) tdm_wait();
#endif
  __syncthreads();

  int buf = 0;
  for (int kc = 0; kc < kend; kc += 32) {
    if (kc + 32 < kend) stage(buf ^ 1, kc + 32);   // DMA next chunk during compute

    // S^T[k,q]: two 16-k tiles, accumulate over d = 128 in four K=32 steps
    v8f st[2];
#pragma unroll
    for (int t = 0; t < 2; ++t) {
      v8f s = {};
#pragma unroll
      for (int c = 0; c < 4; ++c) {
        v16h a = load_frag_a(&Ks[buf][0][0], 128, t * 16, c * 32);
        s = wmma32(a, qb[c], s);
      }
      st[t] = s;
    }

    // scale + causal mask (only diagonal chunks need masking; wave-uniform)
    const bool diag = (kc + 31 > q0);
    const int qg = q0 + (lane & 15);
    const int kb = kc + 8 * (lane >> 4);
    float tmax = -1e30f;
#pragma unroll
    for (int t = 0; t < 2; ++t)
#pragma unroll
      for (int v = 0; v < 8; ++v) {
        float sv = st[t][v] * scale;
        if (diag) {
          const int kg = kb + t * 16 + v;
          sv = (kg <= qg) ? sv : -1e30f;
        }
        st[t][v] = sv;
        tmax = fmaxf(tmax, sv);
      }
    tmax = fmaxf(tmax, __shfl_xor(tmax, 16, 32));
    const float m_new = fmaxf(m_run, tmax);
    const float alpha = __expf(m_run - m_new);
    float rsum = 0.0f;
    const int prow = lane & 15;
    const int pk0  = 8 * (lane >> 4);
#pragma unroll
    for (int t = 0; t < 2; ++t) {
      v8h ph;
#pragma unroll
      for (int v = 0; v < 8; ++v) {
        const float p = __expf(st[t][v] - m_new);
        rsum += p;
        ph[v] = (_Float16)p;
      }
      *(v8h*)&Ps[wave][prow][t * 16 + pk0] = ph;   // contiguous 16B store
    }
    rsum += __shfl_xor(rsum, 16, 32);
    l_run = l_run * alpha + rsum;
    m_run = m_new;
#pragma unroll
    for (int i = 0; i < 8; ++i)
#pragma unroll
      for (int v = 0; v < 8; ++v) ot[i][v] *= alpha;

    // make this wave's P stores visible to its own cross-lane fragment loads
    asm volatile("s_wait_dscnt 0" ::: "memory");

    // O^T[dv,q] += V^T[dv,k] @ P^T[k,q]
    v16h pb = load_frag_b(&Ps[wave][0][0], 32, 0, 0);
#pragma unroll
    for (int i = 0; i < 8; ++i) {
      v16h a = load_frag_a(&Vts[buf][0][0], 32, i * 16, 0);
      ot[i] = wmma32(a, pb, ot[i]);
    }

#if HAVE_TDM
    if (wave == 0) tdm_wait();                   // next tiles landed
#endif
    __syncthreads();
    buf ^= 1;
  }

  // normalize + store O: per-lane dv-run is contiguous -> packed 16B stores
  const float inv_l = 1.0f / l_run;
  const int col = lane & 15, rs = lane >> 4;
  _Float16* orow = O + (size_t)(q0 + col) * 4096 + head * 128;
#pragma unroll
  for (int i = 0; i < 8; ++i) {
    v8h pk;
#pragma unroll
    for (int v = 0; v < 8; ++v) pk[v] = (_Float16)(ot[i][v] * inv_l);
    *(v8h*)&orow[i * 16 + 8 * rs] = pk;
  }
}

// ---------------------------------------------------------------------------
// Host launcher
// ---------------------------------------------------------------------------
extern "C" void kernel_launch(void* const* d_in, const int* in_sizes, int n_in,
                              void* d_out, int out_size, void* d_ws, size_t ws_size,
                              hipStream_t stream) {
  (void)in_sizes; (void)n_in; (void)out_size; (void)ws_size;
  const float* q  = (const float*)d_in[0];
  const float* k  = (const float*)d_in[1];
  const float* v  = (const float*)d_in[2];
  const float* Wq = (const float*)d_in[3];
  const float* Wk = (const float*)d_in[4];
  const float* Wv = (const float*)d_in[5];
  const float* Wd = (const float*)d_in[6];

  const int S = 2048, D = 4096, HD = 4096, KVD = 1024;  // HD = H*DK, KVD = HKV*DK

  char* ws = (char*)d_ws;
  size_t off = 0;
  auto alloc = [&](size_t elems) -> _Float16* {
    _Float16* p = (_Float16*)(ws + off);
    off += ((elems * 2 + 255) / 256) * 256;
    return p;
  };
  _Float16* q16  = alloc((size_t)S * D);
  _Float16* k16  = alloc((size_t)S * D);
  _Float16* v16  = alloc((size_t)S * D);
  _Float16* Wq16 = alloc((size_t)HD * D);
  _Float16* Wk16 = alloc((size_t)KVD * D);
  _Float16* Wv16 = alloc((size_t)KVD * D);
  _Float16* Wd16 = alloc((size_t)D * HD);
  _Float16* Qp   = alloc((size_t)S * HD);
  _Float16* Kp   = alloc((size_t)S * KVD);
  _Float16* VpT  = alloc((size_t)KVD * S);   // V projection stored transposed
  _Float16* O16  = q16;  // q16 is dead after the Q projection; reuse for attn out

  auto cvt = [&](const float* src, _Float16* dst, size_t n) {
    cvt_f32_f16<<<4096, 256, 0, stream>>>(src, dst, (int)n);
  };
  cvt(q,  q16,  (size_t)S * D);
  cvt(k,  k16,  (size_t)S * D);
  cvt(v,  v16,  (size_t)S * D);
  cvt(Wq, Wq16, (size_t)HD * D);
  cvt(Wk, Wk16, (size_t)KVD * D);
  cvt(Wv, Wv16, (size_t)KVD * D);
  cvt(Wd, Wd16, (size_t)D * HD);

  // projections: y = x @ W^T   (V projection written transposed)
  gemm_nt<0><<<dim3(HD / 128,  S / 64), 128, 0, stream>>>(q16, Wq16, Qp,  nullptr, S, HD,  D);
  gemm_nt<0><<<dim3(KVD / 128, S / 64), 128, 0, stream>>>(k16, Wk16, Kp,  nullptr, S, KVD, D);
  gemm_nt<1><<<dim3(KVD / 128, S / 64), 128, 0, stream>>>(v16, Wv16, VpT, nullptr, S, KVD, D);

  // causal GQA attention
  attn_fwd<<<dim3(S / 64, 32), 128, 0, stream>>>(Qp, Kp, VpT, O16);

  // output projection -> f32 d_out
  gemm_nt<2><<<dim3(D / 128, S / 64), 128, 0, stream>>>(O16, Wd16, nullptr, (float*)d_out, S, D, HD);
}
